// PoseODE_88957362634957
// MI455X (gfx1250) — matrix operand order repaired and checked by reading, not compiled, not run on hardware
//
#include <hip/hip_runtime.h>
#include <hip/hip_bf16.h>
#include <cstdint>

// ---------------------------------------------------------------------------
// MI455X / gfx1250 neural-ODE pose regressor.
// GEMMs: v_wmma_f32_16x16x32_bf16 fed from LDS, staged by the CDNA5 async
// global->LDS engine (GLOBAL_LOAD_ASYNC_TO_LDS_B128 / ASYNCcnt), explicit
// depth-2 software pipeline with peeled tails (branch-free steady state).
// ---------------------------------------------------------------------------

typedef __attribute__((ext_vector_type(16))) __bf16 v16bf;
typedef __attribute__((ext_vector_type(8)))  float  v8f;

union Frag {
    v16bf v;
    uint4 u[2];
};

#if __has_builtin(__builtin_amdgcn_s_wait_asynccnt)
#define WAIT_ASYNC(n) __builtin_amdgcn_s_wait_asynccnt(n)
#else
#define WAIT_ASYNC(n) asm volatile("s_wait_asynccnt %0" :: "n"(n) : "memory")
#endif

__device__ __forceinline__ void async_copy_b128(uint32_t lds_byte_addr, const void* gptr) {
    // VDST = LDS byte address (32-bit VGPR), VADDR = 64-bit global address.
    asm volatile("global_load_async_to_lds_b128 %0, %1, off"
                 :: "v"(lds_byte_addr), "v"(gptr) : "memory");
}

__device__ __forceinline__ unsigned short f32_bf16(float f) {
    unsigned int u = __float_as_uint(f);
    u += 0x7FFFu + ((u >> 16) & 1u);     // round-to-nearest-even
    return (unsigned short)(u >> 16);
}

// A-fragment packed offset for element (Mi, Ki) of an MxKdim activation
// matrix.  Layout: [mt][kt][lane][16 bf16] with the ISA 16-bit A layout:
//   lane = (M&15) + 16*hi,  K_local = (s&7) + ((s>>3)<<4) + hi*8
__device__ __forceinline__ size_t a_pack_off(int Mi, int Ki, int Kdim) {
    int mt = Mi >> 4, m = Mi & 15;
    int kt = Ki >> 5, kl = Ki & 31;
    int hi = (kl >> 3) & 1;
    int s  = (kl & 7) | (((kl >> 4) & 1) << 3);
    return ((size_t)(mt * (Kdim >> 5) + kt) * 32 + (m | (hi << 4))) * 16 + s;
}

// ---------------------------------------------------------------------------
// Weight packer: f32 row-major W[K][N] -> bf16 B-fragment tiles
// [nt][kt][lane][16]:  lane = (N&15) + 16*kh,  slot s -> K = (s&7) + kh*8 + ((s>>3)<<4)
// ---------------------------------------------------------------------------
__global__ __launch_bounds__(256)
void pack_weight_kernel(const float* __restrict__ W, unsigned short* __restrict__ Wp,
                        int K, int N) {
    int idx = blockIdx.x * blockDim.x + threadIdx.x;
    if (idx >= K * N) return;
    int s    = idx & 15;
    int lane = (idx >> 4) & 31;
    int rest = idx >> 9;
    int ktn  = K >> 5;
    int kt   = rest % ktn;
    int nt   = rest / ktn;
    int Ks   = kt * 32 + (s & 7) + (lane >> 4) * 8 + ((s >> 3) << 4);
    int Ns   = nt * 16 + (lane & 15);
    Wp[idx] = f32_bf16(W[(size_t)Ks * N + Ns]);
}

// ---------------------------------------------------------------------------
// WMMA GEMM with async-to-LDS, depth-2 double buffering, peeled tails.
//   Block: 256 threads = 8 waves arranged 2(M) x 4(N); wave tile 32x32.
//   Block tile: 64(M) x 128(N); K in 32-deep chunks (ktn >= 4 always here).
//   Per chunk: 4 A-tiles + 8 B-tiles (12 KB) staged with 3 async b128 per
//   thread; each packed tile advances 1024 B per chunk, so steady-state
//   issue is just pointer += 1024.
// ---------------------------------------------------------------------------
__global__ __launch_bounds__(256)
void wmma_gemm_async_kernel(const unsigned short* __restrict__ Ap,
                            const unsigned short* __restrict__ Bp,
                            const float* __restrict__ bias,
                            float* __restrict__ C,
                            int M, int N, int K) {
    __shared__ __align__(16) char smem[2 * 12288];

    const int tid  = (int)threadIdx.x;
    const int lane = tid & 31;
    const int w    = tid >> 5;
    const int wr   = w & 1;        // M position of wave (0..1)
    const int wc   = w >> 1;       // N position of wave (0..3)
    const int bmN  = M >> 6;       // 64-row block tiles
    const int bm   = (int)blockIdx.x % bmN;
    const int bn   = (int)blockIdx.x / bmN;
    const int ktn  = K >> 5;

    const char*    Ab   = (const char*)Ap;
    const char*    Bb   = (const char*)Bp;
    const uint32_t lds0 = (uint32_t)(size_t)(&smem[0]);

    // Per-thread async-copy slots: op0 -> A tile (tq), op1/op2 -> B tiles.
    const int tq  = tid >> 6;             // 0..3
    const int c16 = (tid & 63) << 4;      // 16B chunk within 1KB tile
    const char* s0 = Ab + (((size_t)((bm * 4 + tq)     * ktn)) << 10) + c16;
    const char* s1 = Bb + (((size_t)((bn * 8 + tq)     * ktn)) << 10) + c16;
    const char* s2 = Bb + (((size_t)((bn * 8 + 4 + tq) * ktn)) << 10) + c16;
    const uint32_t d0 = lds0 + (uint32_t)((tq)     << 10) + c16;
    const uint32_t d1 = lds0 + (uint32_t)((4 + tq) << 10) + c16;
    const uint32_t d2 = lds0 + (uint32_t)((8 + tq) << 10) + c16;

    v8f acc[2][2];
#pragma unroll
    for (int i = 0; i < 2; ++i)
#pragma unroll
        for (int j = 0; j < 2; ++j)
#pragma unroll
            for (int r = 0; r < 8; ++r) acc[i][j][r] = 0.0f;

    // Per-wave fragment offsets within a stage buffer.
    const int fa0 = ((wr * 2 + 0) << 10) + lane * 32;
    const int fa1 = ((wr * 2 + 1) << 10) + lane * 32;
    const int fb0 = ((4 + wc * 2 + 0) << 10) + lane * 32;
    const int fb1 = ((4 + wc * 2 + 1) << 10) + lane * 32;

    auto compute = [&](uint32_t boff) {
        const char* base = smem + boff;
        Frag a0, a1, b0, b1;
        const uint4* p;
        p = (const uint4*)(base + fa0); a0.u[0] = p[0]; a0.u[1] = p[1];
        p = (const uint4*)(base + fa1); a1.u[0] = p[0]; a1.u[1] = p[1];
        p = (const uint4*)(base + fb0); b0.u[0] = p[0]; b0.u[1] = p[1];
        p = (const uint4*)(base + fb1); b1.u[0] = p[0]; b1.u[1] = p[1];
        acc[0][0] = __builtin_amdgcn_wmma_f32_16x16x32_bf16(false, a0.v, false, b0.v, (short)0, acc[0][0], false, false);
        acc[0][1] = __builtin_amdgcn_wmma_f32_16x16x32_bf16(false, a0.v, false, b1.v, (short)0, acc[0][1], false, false);
        acc[1][0] = __builtin_amdgcn_wmma_f32_16x16x32_bf16(false, a1.v, false, b0.v, (short)0, acc[1][0], false, false);
        acc[1][1] = __builtin_amdgcn_wmma_f32_16x16x32_bf16(false, a1.v, false, b1.v, (short)0, acc[1][1], false, false);
    };

    // ---- prologue: stage chunk 0 -> buf0, chunk 1 -> buf1 ----
    async_copy_b128(d0, s0);
    async_copy_b128(d1, s1);
    async_copy_b128(d2, s2);
    async_copy_b128(d0 + 12288u, s0 + 1024);
    async_copy_b128(d1 + 12288u, s1 + 1024);
    async_copy_b128(d2 + 12288u, s2 + 1024);
    s0 += 2048; s1 += 2048; s2 += 2048;

    uint32_t boff = 0;
    // ---- steady state: branch-free body ----
#pragma unroll 1
    for (int kt = 0; kt < ktn - 2; ++kt) {
        WAIT_ASYNC(3);                 // retire the oldest stage -> `boff` ready
        __syncthreads();
        compute(boff);
        __syncthreads();               // all waves done reading before refill
        async_copy_b128(d0 + boff, s0);
        async_copy_b128(d1 + boff, s1);
        async_copy_b128(d2 + boff, s2);
        s0 += 1024; s1 += 1024; s2 += 1024;
        boff = 12288u - boff;
    }
    // ---- peeled tails ----
    WAIT_ASYNC(3);
    __syncthreads();
    compute(boff);
    __syncthreads();
    boff = 12288u - boff;
    WAIT_ASYNC(0);
    __syncthreads();
    compute(boff);

    // Epilogue: ISA C/D layout -> row-major f32 (+bias)
    const int nlo = lane & 15;
    const int mhi = (lane >> 4) * 8;
#pragma unroll
    for (int i = 0; i < 2; ++i) {
#pragma unroll
        for (int j = 0; j < 2; ++j) {
            const int Mr0 = (bm * 4 + wr * 2 + i) * 16 + mhi;
            const int Nc  = (bn * 8 + wc * 2 + j) * 16 + nlo;
            const float bv = bias[Nc];
#pragma unroll
            for (int r = 0; r < 8; ++r)
                C[(size_t)(Mr0 + r) * N + Nc] = acc[i][j][r] + bv;
        }
    }
}

// ---------------------------------------------------------------------------
// Activation + pack: P = pack_Afrag(act(H)),  act: 1=tanh, 2=leakyrelu(0.1)
// ---------------------------------------------------------------------------
__global__ __launch_bounds__(256)
void act_pack_kernel(const float* __restrict__ H, unsigned short* __restrict__ P,
                     int Kdim, int total, int act) {
    int idx = blockIdx.x * blockDim.x + threadIdx.x;
    if (idx >= total) return;
    float v = H[idx];
    if (act == 1)      v = tanhf(v);
    else if (act == 2) v = v > 0.0f ? v : 0.1f * v;
    int Mi = idx / Kdim, Ki = idx - Mi * Kdim;
    P[a_pack_off(Mi, Ki, Kdim)] = f32_bf16(v);
}

// ---------------------------------------------------------------------------
// RK4 (3/8 rule) state combines; dt = ts[step+1]-ts[step] read on device.
//   mode 1: X = y + dt/3 * k1
//   mode 2: X = y + dt*(k2 - k1/3)
//   mode 3: X = y + dt*(k1 - k2 + k3)
//   mode 4: y += dt*0.125*(k1 + 3(k2+k3) + k4)  (writes y f32 AND packed y)
// ---------------------------------------------------------------------------
__global__ __launch_bounds__(256)
void rk_combine_kernel(float* __restrict__ y,
                       const float* __restrict__ k1, const float* __restrict__ k2,
                       const float* __restrict__ k3, const float* __restrict__ k4,
                       const float* __restrict__ ts, int step, int mode,
                       unsigned short* __restrict__ P) {
    int idx = blockIdx.x * blockDim.x + threadIdx.x;
    if (idx >= 256 * 768) return;
    const float dt = ts[step + 1] - ts[step];
    float v;
    if (mode == 1)      v = y[idx] + dt * (1.0f / 3.0f) * k1[idx];
    else if (mode == 2) v = y[idx] + dt * (k2[idx] - (1.0f / 3.0f) * k1[idx]);
    else if (mode == 3) v = y[idx] + dt * (k1[idx] - k2[idx] + k3[idx]);
    else {
        v = y[idx] + dt * 0.125f * (k1[idx] + 3.0f * (k2[idx] + k3[idx]) + k4[idx]);
        y[idx] = v;
    }
    int Mi = idx / 768, Ki = idx - Mi * 768;
    P[a_pack_off(Mi, Ki, 768)] = f32_bf16(v);
}

// ---------------------------------------------------------------------------
// Final regressor layer: pose[m,n] = b4[n] + sum_k lrelu(H3[m,k]) * w4[k,n]
// ---------------------------------------------------------------------------
__global__ __launch_bounds__(256)
void reg_final_kernel(const float* __restrict__ H3, const float* __restrict__ w4,
                      const float* __restrict__ b4, float* __restrict__ out) {
    int idx = blockIdx.x * blockDim.x + threadIdx.x;
    if (idx >= 256 * 6) return;
    int m = idx / 6, n = idx - m * 6;
    float acc = b4[n];
    for (int k = 0; k < 128; ++k) {
        float h = H3[m * 128 + k];
        h = h > 0.0f ? h : 0.1f * h;
        acc += h * w4[k * 6 + n];
    }
    out[idx] = acc;
}

// ---------------------------------------------------------------------------
// Host launcher
// ---------------------------------------------------------------------------
extern "C" void kernel_launch(void* const* d_in, const int* in_sizes, int n_in,
                              void* d_out, int out_size, void* d_ws, size_t ws_size,
                              hipStream_t stream) {
    (void)in_sizes; (void)n_in; (void)out_size; (void)ws_size;
    const float* ts  = (const float*)d_in[4];
    const float* ow1 = (const float*)d_in[5];
    const float* ob1 = (const float*)d_in[6];
    const float* ow2 = (const float*)d_in[7];
    const float* ob2 = (const float*)d_in[8];
    const float* rw1 = (const float*)d_in[9];
    const float* rb1 = (const float*)d_in[10];
    const float* rw2 = (const float*)d_in[11];
    const float* rb2 = (const float*)d_in[12];
    const float* rw3 = (const float*)d_in[13];
    const float* rb3 = (const float*)d_in[14];
    const float* rw4 = (const float*)d_in[15];
    const float* rb4 = (const float*)d_in[16];

    char* ws = (char*)d_ws;
    size_t off = 0;
    auto alloc = [&](size_t bytes) {
        void* p = ws + off;
        off = (off + bytes + 255) & ~(size_t)255;
        return p;
    };
    // Packed bf16 weights
    unsigned short* w1p  = (unsigned short*)alloc(768 * 768 * 2);
    unsigned short* w2p  = (unsigned short*)alloc(768 * 768 * 2);
    unsigned short* rw1p = (unsigned short*)alloc(768 * 128 * 2);
    unsigned short* rw2p = (unsigned short*)alloc(128 * 256 * 2);
    unsigned short* rw3p = (unsigned short*)alloc(256 * 128 * 2);
    // f32 state / intermediates
    float* y  = (float*)alloc(256 * 768 * 4);
    float* H  = (float*)alloc(256 * 768 * 4);
    float* k1 = (float*)alloc(256 * 768 * 4);
    float* k2 = (float*)alloc(256 * 768 * 4);
    float* k3 = (float*)alloc(256 * 768 * 4);
    float* k4 = (float*)alloc(256 * 768 * 4);
    float* H1 = (float*)alloc(256 * 128 * 4);
    float* H2 = (float*)alloc(256 * 256 * 4);
    float* H3 = (float*)alloc(256 * 128 * 4);
    // packed bf16 activations
    unsigned short* yp  = (unsigned short*)alloc(256 * 768 * 2);
    unsigned short* Xp  = (unsigned short*)alloc(256 * 768 * 2);
    unsigned short* Tp  = (unsigned short*)alloc(256 * 768 * 2);
    unsigned short* H1p = (unsigned short*)alloc(256 * 128 * 2);
    unsigned short* H2p = (unsigned short*)alloc(256 * 256 * 2);

    auto pack_w = [&](const float* W, unsigned short* Wp, int K, int N) {
        int total = K * N;
        pack_weight_kernel<<<(total + 255) / 256, 256, 0, stream>>>(W, Wp, K, N);
    };
    auto gemm = [&](const unsigned short* Ap, const unsigned short* Bp,
                    const float* bias, float* C, int M, int N, int K) {
        int blocks = (M >> 6) * (N >> 7);    // 64x128 block tiles
        wmma_gemm_async_kernel<<<blocks, 256, 0, stream>>>(Ap, Bp, bias, C, M, N, K);
    };
    auto actp = [&](const float* Hs, unsigned short* P, int Kdim, int act) {
        int total = 256 * Kdim;
        act_pack_kernel<<<(total + 255) / 256, 256, 0, stream>>>(Hs, P, Kdim, total, act);
    };
    auto comb = [&](int step, int mode, unsigned short* P) {
        rk_combine_kernel<<<(256 * 768 + 255) / 256, 256, 0, stream>>>(
            y, k1, k2, k3, k4, ts, step, mode, P);
    };
    auto ode_f = [&](const unsigned short* Ain, float* kout) {
        gemm(Ain, w1p, ob1, H, 256, 768, 768);
        actp(H, Tp, 768, /*tanh*/1);
        gemm(Tp, w2p, ob2, kout, 256, 768, 768);
    };

    // ---- prep (re-done every call; deterministic) ----
    pack_w(ow1, w1p, 768, 768);
    pack_w(ow2, w2p, 768, 768);
    pack_w(rw1, rw1p, 768, 128);
    pack_w(rw2, rw2p, 128, 256);
    pack_w(rw3, rw3p, 256, 128);
    hipMemsetAsync(y,  0, 256 * 768 * 4, stream);
    hipMemsetAsync(yp, 0, 256 * 768 * 2, stream);

    // ---- 64 sequential RK4 (3/8 rule) steps ----
    for (int i = 0; i < 64; ++i) {
        ode_f(yp, k1);
        comb(i, 1, Xp);
        ode_f(Xp, k2);
        comb(i, 2, Xp);
        ode_f(Xp, k3);
        comb(i, 3, Xp);
        ode_f(Xp, k4);
        comb(i, 4, yp);   // updates y (f32) and yp (packed)
    }

    // ---- regressor: 768 -> 128 -> 256 -> 128 -> 6 with leaky ReLU ----
    gemm(yp, rw1p, rb1, H1, 256, 128, 768);
    actp(H1, H1p, 128, /*lrelu*/2);
    gemm(H1p, rw2p, rb2, H2, 256, 256, 128);
    actp(H2, H2p, 256, /*lrelu*/2);
    gemm(H2p, rw3p, rb3, H3, 256, 128, 256);

    float* out = (float*)d_out;
    reg_final_kernel<<<(256 * 6 + 255) / 256, 256, 0, stream>>>(H3, rw4, rb4, out);
    // second output: yT [256,768] f32, concatenated after pose [256,1,6]
    hipMemcpyAsync(out + 256 * 6, y, 256 * 768 * 4, hipMemcpyDeviceToDevice, stream);
}
